// StructuredLSTMCell_24790551232936
// MI455X (gfx1250) — matrix operand order repaired and checked by reading, not compiled, not run on hardware
//
#include <hip/hip_runtime.h>

#define B_DIM 4096
#define PD    1024
#define HD    2048
#define XD    1024
#define K_TOT (PD + HD + XD)   // 4096

#define BM 128
#define BN 64
#define BK 32
#define NSTEP (K_TOT / BK)     // 128

typedef __attribute__((ext_vector_type(16))) __bf16 v16bf;
typedef __attribute__((ext_vector_type(8)))  __bf16 v8bf;
typedef __attribute__((ext_vector_type(8)))  float  v8f;
typedef __attribute__((ext_vector_type(4)))  float  v4f;   // POD float4 (AS-cast friendly)

// Force addrspace(1): loads lower to global_load_* (LOADcnt only, no DScnt coupling)
__device__ __forceinline__ v4f gload4(const float* p) {
  return *(const __attribute__((address_space(1))) v4f*)(p);
}
__device__ __forceinline__ float gload1(const float* p) {
  return *(const __attribute__((address_space(1))) float*)(p);
}
__device__ __forceinline__ void gstore1(float* p, float v) {
  *(__attribute__((address_space(1))) float*)(p) = v;
}

__device__ __forceinline__ v8bf pack8(v4f a, v4f b) {
  v8bf h;
  h[0] = (__bf16)a.x; h[1] = (__bf16)a.y; h[2] = (__bf16)a.z; h[3] = (__bf16)a.w;
  h[4] = (__bf16)b.x; h[5] = (__bf16)b.y; h[6] = (__bf16)b.z; h[7] = (__bf16)b.w;
  return h;
}

__device__ __forceinline__ float fast_sigmoid(float x) {
  return 1.0f / (1.0f + __expf(-x));
}
__device__ __forceinline__ float fast_tanh(float x) {
  return 1.0f - 2.0f / (__expf(2.0f * x) + 1.0f);
}

__global__ __launch_bounds__(256)
void lstm_fused_wmma(
    const float* __restrict__ Pm,  const float* __restrict__ hid,
    const float* __restrict__ cel, const float* __restrict__ ext,
    const float* __restrict__ Wpi, const float* __restrict__ bpi,
    const float* __restrict__ Wpf, const float* __restrict__ bpf,
    const float* __restrict__ Wpg, const float* __restrict__ bpg,
    const float* __restrict__ Wpo, const float* __restrict__ bpo,
    const float* __restrict__ Whi, const float* __restrict__ Whf,
    const float* __restrict__ Whg, const float* __restrict__ Who,
    const float* __restrict__ Wxi, const float* __restrict__ Wxf,
    const float* __restrict__ Wxg, const float* __restrict__ Wxo,
    float* __restrict__ h_out, float* __restrict__ c_out)
{
  __shared__ __bf16 lsA[2][BM * BK];        // 2 x 8 KB
  __shared__ __bf16 lsW[2][4 * BN * BK];    // 2 x 16 KB

  const int tid   = threadIdx.x;
  const int lane  = tid & 31;
  const int wave  = tid >> 5;
  const int waveM = wave & 3;               // 4 slabs of 32 rows
  const int waveN = wave >> 2;              // 2 slabs of 32 cols
  const int half  = lane >> 4;
  const int l15   = lane & 15;

  const int m0 = blockIdx.x * BM;
  const int n0 = blockIdx.y * BN;

  // ---- staging map: each thread owns 8-contiguous-float chunks
  //   A tile 128x32: chunk c -> row = c>>2, col = (c&3)*8 ; thread owns c = tid and c = tid+256
  //   W tiles 4x(64x32): per gate 256 chunks; thread owns chunk tid of each gate (gate == q)
  const int rA  = tid >> 2;                 // 0..63
  const int c8  = (tid & 3) * 8;            // 0,8,16,24
  // per-thread 32-bit element offsets, strides folded as compile-time shifts
  const int baseA1 = (m0 + rA) * 1024 + c8;   // stride-1024 segments (P / ext)
  const int baseA2 = (m0 + rA) * 2048 + c8;   // stride-2048 segment (hidden)
  const int baseW1 = (n0 + rA) * 1024 + c8;
  const int baseW2 = (n0 + rA) * 2048 + c8;

  v8f acc[4][2][2] = {};                    // [gate][mtile][ntile]

  v4f ra[4];                                // A staging: 2 chunks x 2 float4
  v4f rw[8];                                // W staging: 4 gates x 2 float4

  // ---- issue all 12 global loads for k-step `ks` (no intervening waits)
  auto stage_load = [&](int ks) {
    if (ks < PD) {
      const int oa = baseA1 + ks;
      ra[0] = gload4(Pm + oa);                 ra[1] = gload4(Pm + oa + 4);
      ra[2] = gload4(Pm + oa + 64 * 1024);     ra[3] = gload4(Pm + oa + 64 * 1024 + 4);
      const int ow = baseW1 + ks;
      rw[0] = gload4(Wpi + ow);  rw[1] = gload4(Wpi + ow + 4);
      rw[2] = gload4(Wpf + ow);  rw[3] = gload4(Wpf + ow + 4);
      rw[4] = gload4(Wpg + ow);  rw[5] = gload4(Wpg + ow + 4);
      rw[6] = gload4(Wpo + ow);  rw[7] = gload4(Wpo + ow + 4);
    } else if (ks < PD + HD) {
      const int kk = ks - PD;
      const int oa = baseA2 + kk;
      ra[0] = gload4(hid + oa);                ra[1] = gload4(hid + oa + 4);
      ra[2] = gload4(hid + oa + 64 * 2048);    ra[3] = gload4(hid + oa + 64 * 2048 + 4);
      const int ow = baseW2 + kk;
      rw[0] = gload4(Whi + ow);  rw[1] = gload4(Whi + ow + 4);
      rw[2] = gload4(Whf + ow);  rw[3] = gload4(Whf + ow + 4);
      rw[4] = gload4(Whg + ow);  rw[5] = gload4(Whg + ow + 4);
      rw[6] = gload4(Who + ow);  rw[7] = gload4(Who + ow + 4);
    } else {
      const int kk = ks - (PD + HD);
      const int oa = baseA1 + kk;
      ra[0] = gload4(ext + oa);                ra[1] = gload4(ext + oa + 4);
      ra[2] = gload4(ext + oa + 64 * 1024);    ra[3] = gload4(ext + oa + 64 * 1024 + 4);
      const int ow = baseW1 + kk;
      rw[0] = gload4(Wxi + ow);  rw[1] = gload4(Wxi + ow + 4);
      rw[2] = gload4(Wxf + ow);  rw[3] = gload4(Wxf + ow + 4);
      rw[4] = gload4(Wxg + ow);  rw[5] = gload4(Wxg + ow + 4);
      rw[6] = gload4(Wxo + ow);  rw[7] = gload4(Wxo + ow + 4);
    }
  };

  // ---- convert + 16-byte LDS stores (6 x ds_store_b128 per thread)
  auto stage_store = [&](int buf) {
    __bf16* pA = &lsA[buf][rA * BK + c8];
    *(v8bf*)(pA)            = pack8(ra[0], ra[1]);
    *(v8bf*)(pA + 64 * BK)  = pack8(ra[2], ra[3]);
    __bf16* pW = &lsW[buf][rA * BK + c8];
    *(v8bf*)(pW + 0 * (BN * BK)) = pack8(rw[0], rw[1]);
    *(v8bf*)(pW + 1 * (BN * BK)) = pack8(rw[2], rw[3]);
    *(v8bf*)(pW + 2 * (BN * BK)) = pack8(rw[4], rw[5]);
    *(v8bf*)(pW + 3 * (BN * BK)) = pack8(rw[6], rw[7]);
  };

  // ---- prologue: fill buffer 0
  stage_load(0);
  stage_store(0);
  __syncthreads();

  for (int t = 0; t < NSTEP; ++t) {
    const int buf = t & 1;

    // prefetch next tile's global data while this tile computes
    if (t + 1 < NSTEP) stage_load((t + 1) * BK);

    // ---- A fragments (16x32 bf16, contiguous K: lane<16 -> K 0-15; lane>=16 -> K 16-31)
    v16bf af[2];
    #pragma unroll
    for (int mt = 0; mt < 2; ++mt) {
      const __bf16* pr = &lsA[buf][(waveM * 32 + mt * 16 + l15) * BK];
      v8bf lo = *(const v8bf*)(pr + half * 16);
      v8bf hi = *(const v8bf*)(pr + half * 16 + 8);
      #pragma unroll
      for (int i = 0; i < 8; ++i) { af[mt][i] = lo[i]; af[mt][8 + i] = hi[i]; }
    }

    // ---- B fragments + 16 WMMAs
    #pragma unroll
    for (int g = 0; g < 4; ++g) {
      #pragma unroll
      for (int nt = 0; nt < 2; ++nt) {
        const __bf16* pc = &lsW[buf][g * (BN * BK) + (waveN * 32 + nt * 16 + l15) * BK + half * 16];
        v16bf bfr = *(const v16bf*)pc;
        #pragma unroll
        for (int mt = 0; mt < 2; ++mt) {
          acc[g][mt][nt] = __builtin_amdgcn_wmma_f32_16x16x32_bf16(
              false, af[mt], false, bfr, (short)0, acc[g][mt][nt], false, false);
        }
      }
    }

    // write next tile into the other buffer, then one barrier per iteration
    if (t + 1 < NSTEP) stage_store((t + 1) & 1);
    __syncthreads();
  }

  // ---- fused LSTM epilogue directly on accumulators
  const int mBase = m0 + waveM * 32;
  const int nBase = n0 + waveN * 32;
  #pragma unroll
  for (int nt = 0; nt < 2; ++nt) {
    const int n = nBase + nt * 16 + l15;
    const float bi = gload1(bpi + n), bf_ = gload1(bpf + n);
    const float bg = gload1(bpg + n), bo = gload1(bpo + n);
    #pragma unroll
    for (int mt = 0; mt < 2; ++mt) {
      #pragma unroll
      for (int v = 0; v < 8; ++v) {
        const int m = mBase + mt * 16 + half * 8 + v;   // C/D layout: VGPR v -> M=v (+8 hi lanes)
        const float gi = acc[0][mt][nt][v] + bi;
        const float gf = acc[1][mt][nt][v] + bf_;
        const float gg = acc[2][mt][nt][v] + bg;
        const float go = acc[3][mt][nt][v] + bo;
        const float iv = fast_sigmoid(gi);
        const float fv = fast_sigmoid(gf);
        const float gv = fast_tanh(gg);
        const float ov = fast_sigmoid(go);
        const size_t idx = (size_t)m * HD + n;
        const float c = fv * gload1(cel + idx) + iv * gv;
        const float h = ov * fast_tanh(c);
        gstore1(c_out + idx, c);
        gstore1(h_out + idx, h);
      }
    }
  }
}

extern "C" void kernel_launch(void* const* d_in, const int* in_sizes, int n_in,
                              void* d_out, int out_size, void* d_ws, size_t ws_size,
                              hipStream_t stream) {
  const float* Pm  = (const float*)d_in[0];
  const float* hid = (const float*)d_in[1];
  const float* cel = (const float*)d_in[2];
  const float* ext = (const float*)d_in[3];
  const float* Wpi = (const float*)d_in[4];  const float* bpi = (const float*)d_in[5];
  const float* Wpf = (const float*)d_in[6];  const float* bpf = (const float*)d_in[7];
  const float* Wpg = (const float*)d_in[8];  const float* bpg = (const float*)d_in[9];
  const float* Wpo = (const float*)d_in[10]; const float* bpo = (const float*)d_in[11];
  const float* Whi = (const float*)d_in[12]; const float* Whf = (const float*)d_in[13];
  const float* Whg = (const float*)d_in[14]; const float* Who = (const float*)d_in[15];
  const float* Wxi = (const float*)d_in[16]; const float* Wxf = (const float*)d_in[17];
  const float* Wxg = (const float*)d_in[18]; const float* Wxo = (const float*)d_in[19];

  float* h_out = (float*)d_out;
  float* c_out = h_out + (size_t)B_DIM * HD;

  dim3 grid(B_DIM / BM, HD / BN);   // 32 x 32 = 1024 blocks
  dim3 block(256);                  // 8 wave32 waves
  lstm_fused_wmma<<<grid, block, 0, stream>>>(
      Pm, hid, cel, ext,
      Wpi, bpi, Wpf, bpf, Wpg, bpg, Wpo, bpo,
      Whi, Whf, Whg, Who,
      Wxi, Wxf, Wxg, Wxo,
      h_out, c_out);
}
